// MorphologicalLayer_74019466379884
// MI455X (gfx1250) — matrix operand order repaired and checked by reading, not compiled, not run on hardware
//
#include <hip/hip_runtime.h>
#include <stdint.h>

#define BH   16
#define IMH  512
#define IMW  512
#define HW   (IMH*IMW)
#define NPIX (BH*HW)
#define INF_LAB 0x40000000
#define NITER  48
#define NSWEEP (2*NITER)
#define TPB 256

typedef __attribute__((ext_vector_type(16))) _Float16 v16h;
typedef __attribute__((ext_vector_type(8)))  float    v8f;

__host__ __device__ constexpr int disk_count(int R) {
    int c = 0;
    for (int dy = -R; dy <= R; ++dy)
        for (int dx = -R; dx <= R; ++dx)
            if (dy*dy + dx*dx <= R*R) ++c;
    return c;
}

// ---------------------------------------------------------------------------
// Morphology (erode/dilate with disk(R)) as WMMA GEMM over 16x16 output tiles.
//   D[m][n] = sum_{dy, cx} in(y0+m+dy, x0+cx-R) * se(dy, cx-R-n)
// K is enumerated as chunks of 32: chunk c <-> dy = c-R, localK <-> halo col cx.
// A halo tile ((16+2R) rows x 32 cols, f16) is staged in LDS once per wave.
// One wave (32 lanes) per 16x16 output tile; EXEC all-ones around the WMMAs.
// ---------------------------------------------------------------------------
template<int R, bool ERODE>
__global__ void morph_kernel(const uint8_t* __restrict__ in, uint8_t* __restrict__ out) {
    constexpr int NCH    = 2*R + 1;      // one wmma chunk per dy row
    constexpr int TROWS  = 16 + 2*R;     // halo rows staged in LDS
    constexpr int KCP    = 34;           // LDS row stride in halfs (bank spread)
    constexpr int NSE    = disk_count(R);

    __shared__ _Float16 tile[TROWS * KCP];

    const int lane = threadIdx.x;        // 0..31
    const int x0 = blockIdx.x * 16;
    const int y0 = blockIdx.y * 16;
    const uint8_t* src = in  + (size_t)blockIdx.z * HW;
    uint8_t*       dst = out + (size_t)blockIdx.z * HW;
    const float padv = ERODE ? 1.0f : 0.0f;

    // ---- stage halo tile into LDS: branch-free clamped loads + select ----
#pragma unroll
    for (int it = 0; it < TROWS; ++it) {
        const int yy = it;
        const int xx = lane;                       // 32 cols per row
        const int y = y0 + yy - R;
        const int x = x0 + xx - R;
        const int inb = ((unsigned)y < (unsigned)IMH) & ((unsigned)x < (unsigned)IMW);
        const int yc = y < 0 ? 0 : (y > IMH-1 ? IMH-1 : y);
        const int xc = x < 0 ? 0 : (x > IMW-1 ? IMW-1 : x);
        const float fv = (float)src[yc*IMW + xc];  // unconditional load
        tile[yy*KCP + xx] = (_Float16)(inb ? fv : padv);
    }
    __syncthreads();

    const int m   = lane & 15;              // A row M == B col N
    const int hiA = (lane & 16) >> 1;       // +8 local-K offset (A layout, upper lanes)
    const int hiB = (lane & 16);            // unused; B uses the same +8 offset (hiA) as A
    const _Float16* arow = tile + m*KCP + hiA;
    const int t = m + R - hiA;              // B weight: nonzero iff (j+(j&8)-t)^2 <= R^2-dy^2

    v8f acc = {};
#pragma unroll
    for (int c = 0; c < NCH; ++c) {
        const int dy = c - R;               // constant after unroll
        const int rad2 = R*R - dy*dy;
        v16h a, b;
#pragma unroll
        for (int j = 0; j < 16; ++j) {
            // A: localK = j + (j&8) + hiA ; address = (m+c)*KCP + localK
            a[j] = arow[c*KCP + j + (j & 8)];
            // B: localK = j + (j&8) + hiA ; dx = localK - R - m = j + (j&8) - t
            const int d = j + (j & 8) - t;
            b[j] = (d*d <= rad2) ? (_Float16)1.0f : (_Float16)0.0f;
        }
        acc = __builtin_amdgcn_wmma_f32_16x16x32_f16(
                  false, a, false, b, (short)0, acc, false, false);
    }

    const float thr = ERODE ? ((float)NSE - 0.5f) : 0.5f;
    const int mbase = (lane & 16) ? 8 : 0;  // C/D layout: lanes 16-31 hold M=8..15
#pragma unroll
    for (int v = 0; v < 8; ++v) {
        const int y = y0 + mbase + v;
        const int x = x0 + m;
        dst[y*IMW + x] = (acc[v] >= thr) ? (uint8_t)1 : (uint8_t)0;
    }
}

// ---------------------------------------------------------------------------
// Elementwise helpers
// ---------------------------------------------------------------------------
__global__ void threshold_kernel(const float* __restrict__ in, uint8_t* __restrict__ out) {
    int i = blockIdx.x * blockDim.x + threadIdx.x;
    if (i < NPIX) out[i] = (in[i] > 0.0f) ? 1 : 0;
}

__global__ void to_float_kernel(const uint8_t* __restrict__ m, float* __restrict__ out) {
    int i = blockIdx.x * blockDim.x + threadIdx.x;
    if (i < NPIX) out[i] = (float)m[i];
}

__global__ void zero_i32(int* p, int n) {
    int i = blockIdx.x * blockDim.x + threadIdx.x;
    if (i < n) p[i] = 0;
}

__global__ void init_flags_kernel(int* flags, int n) {
    int i = blockIdx.x * blockDim.x + threadIdx.x;
    if (i < n) flags[i] = (i < 2) ? 1 : 0;
}

// ---------------------------------------------------------------------------
// Connected-component labeling: min-label propagation via segmented cummin
// ---------------------------------------------------------------------------
__global__ void init_labels_kernel(const uint8_t* __restrict__ mask, int invert,
                                   int* __restrict__ lab) {
    int i = blockIdx.x * blockDim.x + threadIdx.x;
    if (i >= NPIX) return;
    int fg = mask[i]; if (invert) fg = !fg;
    lab[i] = fg ? (i % HW) : INF_LAB;
}

__global__ void row_sweep_kernel(const uint8_t* __restrict__ mask, int invert,
                                 int* __restrict__ lab, int* __restrict__ flags, int s) {
    if (!(flags[s] | flags[s + 1])) return;   // converged: no change in last 2 sweeps
    int t = blockIdx.x * blockDim.x + threadIdx.x;
    if (t >= BH * IMH) return;
    int img = t / IMH, y = t % IMH;
    const uint8_t* mrow = mask + (size_t)img*HW + (size_t)y*IMW;
    int* lrow = lab + (size_t)img*HW + (size_t)y*IMW;
    int changed = 0;
    int run = INF_LAB;
    for (int x = 0; x < IMW; ++x) {
        int fg = mrow[x]; if (invert) fg = !fg;
        if (fg) {
            int lv = lrow[x];
            if (lv < run) run = lv;
            if (run != lv) { lrow[x] = run; changed = 1; }
        } else run = INF_LAB;
    }
    run = INF_LAB;
    for (int x = IMW - 1; x >= 0; --x) {
        int fg = mrow[x]; if (invert) fg = !fg;
        if (fg) {
            int lv = lrow[x];
            if (lv < run) run = lv;
            if (run != lv) { lrow[x] = run; changed = 1; }
        } else run = INF_LAB;
    }
    if (changed) flags[s + 2] = 1;
}

__global__ void col_sweep_kernel(const uint8_t* __restrict__ mask, int invert,
                                 int* __restrict__ lab, int* __restrict__ flags, int s) {
    if (!(flags[s] | flags[s + 1])) return;
    int t = blockIdx.x * blockDim.x + threadIdx.x;
    if (t >= BH * IMW) return;
    int img = t / IMW, x = t % IMW;
    const uint8_t* mcol = mask + (size_t)img*HW + x;
    int* lcol = lab + (size_t)img*HW + x;
    int changed = 0;
    int run = INF_LAB;
    for (int y = 0; y < IMH; ++y) {
        int fg = mcol[(size_t)y*IMW]; if (invert) fg = !fg;
        if (fg) {
            int lv = lcol[(size_t)y*IMW];
            if (lv < run) run = lv;
            if (run != lv) { lcol[(size_t)y*IMW] = run; changed = 1; }
        } else run = INF_LAB;
    }
    run = INF_LAB;
    for (int y = IMH - 1; y >= 0; --y) {
        int fg = mcol[(size_t)y*IMW]; if (invert) fg = !fg;
        if (fg) {
            int lv = lcol[(size_t)y*IMW];
            if (lv < run) run = lv;
            if (run != lv) { lcol[(size_t)y*IMW] = run; changed = 1; }
        } else run = INF_LAB;
    }
    if (changed) flags[s + 2] = 1;
}

__global__ void accum_sizes_kernel(const uint8_t* __restrict__ mask, int invert,
                                   const int* __restrict__ lab, int* __restrict__ sizes) {
    int i = blockIdx.x * blockDim.x + threadIdx.x;
    if (i >= NPIX) return;
    int fg = mask[i]; if (invert) fg = !fg;
    if (fg) {
        int img = i / HW;
        atomicAdd(&sizes[(size_t)img*HW + lab[i]], 1);
    }
}

__global__ void count_comp_kernel(const int* __restrict__ sizes, int* __restrict__ ncomp) {
    int i = blockIdx.x * blockDim.x + threadIdx.x;
    if (i >= NPIX) return;
    if (sizes[i] > 0) atomicAdd(&ncomp[i / HW], 1);
}

// holes==0: out = guard ? (ncomp>1 ? kept : fg) : kept     (in CCL orientation)
// holes==1: out = !kept   (back in original orientation: fills small bg comps)
__global__ void filter_kernel(uint8_t* __restrict__ mask, int invert,
                              const int* __restrict__ lab, const int* __restrict__ sizes,
                              const int* __restrict__ ncomp, int min_size,
                              int guard, int holes) {
    int i = blockIdx.x * blockDim.x + threadIdx.x;
    if (i >= NPIX) return;
    int mv = mask[i];
    int fg = invert ? !mv : mv;
    int img = i / HW;
    int kept = 0;
    if (fg) kept = (sizes[(size_t)img*HW + lab[i]] >= min_size) ? 1 : 0;
    int res;
    if (holes) res = !kept;
    else       res = guard ? ((ncomp[img] > 1) ? kept : fg) : kept;
    mask[i] = (uint8_t)res;
}

// ---------------------------------------------------------------------------
// Host-side sequencing
// ---------------------------------------------------------------------------
static void run_ccl(uint8_t* mask, int invert, int min_size, int guard, int holes,
                    int* lab, int* sizes, int* ncomp, int* flags, hipStream_t stream) {
    const int pixBlocks  = (NPIX + TPB - 1) / TPB;
    const int rowBlocks  = (BH*IMH + TPB - 1) / TPB;
    const int colBlocks  = (BH*IMW + TPB - 1) / TPB;
    init_flags_kernel<<<1, 128, 0, stream>>>(flags, NSWEEP + 2);
    init_labels_kernel<<<pixBlocks, TPB, 0, stream>>>(mask, invert, lab);
    for (int it = 0; it < NITER; ++it) {
        int s = 2 * it;
        row_sweep_kernel<<<rowBlocks, TPB, 0, stream>>>(mask, invert, lab, flags, s);
        col_sweep_kernel<<<colBlocks, TPB, 0, stream>>>(mask, invert, lab, flags, s + 1);
    }
    zero_i32<<<pixBlocks, TPB, 0, stream>>>(sizes, NPIX);
    zero_i32<<<1, 64, 0, stream>>>(ncomp, BH);
    accum_sizes_kernel<<<pixBlocks, TPB, 0, stream>>>(mask, invert, lab, sizes);
    count_comp_kernel<<<pixBlocks, TPB, 0, stream>>>(sizes, ncomp);
    filter_kernel<<<pixBlocks, TPB, 0, stream>>>(mask, invert, lab, sizes, ncomp,
                                                 min_size, guard, holes);
}

extern "C" void kernel_launch(void* const* d_in, const int* in_sizes, int n_in,
                              void* d_out, int out_size, void* d_ws, size_t ws_size,
                              hipStream_t stream) {
    (void)in_sizes; (void)n_in; (void)out_size; (void)ws_size;
    const float* in  = (const float*)d_in[0];
    float*       out = (float*)d_out;

    char* ws = (char*)d_ws;
    size_t off = 0;
    auto carve = [&](size_t bytes) -> char* {
        char* p = ws + off;
        off = (off + bytes + 255) & ~(size_t)255;
        return p;
    };
    uint8_t* maskA = (uint8_t*)carve(NPIX);
    uint8_t* maskB = (uint8_t*)carve(NPIX);
    int*     lab   = (int*)carve((size_t)NPIX * sizeof(int));
    int*     sizes = (int*)carve((size_t)NPIX * sizeof(int));
    int*     ncomp = (int*)carve(BH * sizeof(int));
    int*     flags = (int*)carve((NSWEEP + 2) * sizeof(int));

    const int pixBlocks = (NPIX + TPB - 1) / TPB;
    const dim3 tileGrid(IMW / 16, IMH / 16, BH);

    // fg = mask > 0
    threshold_kernel<<<pixBlocks, TPB, 0, stream>>>(in, maskA);
    // remove_small_objects(fg, 2000, guard=True)
    run_ccl(maskA, 0, 2000, 1, 0, lab, sizes, ncomp, flags, stream);
    // fill small holes: ~remove_small_objects(~fg, 301, guard=False)
    run_ccl(maskA, 1, 301, 0, 1, lab, sizes, ncomp, flags, stream);
    // erode disk(2)
    morph_kernel<2, true ><<<tileGrid, 32, 0, stream>>>(maskA, maskB);
    // opening disk(5): erode then dilate
    morph_kernel<5, true ><<<tileGrid, 32, 0, stream>>>(maskB, maskA);
    morph_kernel<5, false><<<tileGrid, 32, 0, stream>>>(maskA, maskB);
    // remove_small_objects(fg, 2000, guard=True)
    run_ccl(maskB, 0, 2000, 1, 0, lab, sizes, ncomp, flags, stream);
    // dilate disk(1)
    morph_kernel<1, false><<<tileGrid, 32, 0, stream>>>(maskB, maskA);
    // cast to float
    to_float_kernel<<<pixBlocks, TPB, 0, stream>>>(maskA, out);
}
